// UNetEncoderBlock_25598005084690
// MI455X (gfx1250) — matrix operand order repaired and checked by reading, not compiled, not run on hardware
//
#include <hip/hip_runtime.h>

typedef __attribute__((ext_vector_type(16))) __bf16 v16bf;
typedef __attribute__((ext_vector_type(8)))  __bf16 v8bf;
typedef __attribute__((ext_vector_type(4)))  __bf16 v4bf;
typedef __attribute__((ext_vector_type(8)))  float  v8f;
typedef __attribute__((ext_vector_type(4)))  float  v4f;

#define NPTS 400000
#define NCH  64
#define KOFF 27
#define NELEM (NPTS * NCH)

__device__ __forceinline__ v8f wmma_bf16(v16bf a, v16bf b, v8f c) {
    // (neg_a, A, neg_b, B, c_mod, C, reuse_a, reuse_b)
    return __builtin_amdgcn_wmma_f32_16x16x32_bf16(false, a, false, b, (short)0, c, false, false);
}

// ---------------------------------------------------------------------------
// Pre-swizzle W[k] (64x64 f32, Cin-major) into WMMA B-fragment layout (bf16).
// Fragment id = (k*2+q)*4+nt covers K rows 32q..32q+31, N cols 16nt..16nt+15.
// B 32x16 bf16 in this code's k order: lane = n + 16*(kk/16), half h = kk%16.
// ---------------------------------------------------------------------------
__global__ void swizzle_w(const float* __restrict__ W, __bf16* __restrict__ wb) {
    int t = blockIdx.x * blockDim.x + threadIdx.x;      // 27*8*32 = 6912 threads
    if (t >= KOFF * 8 * 32) return;
    int lane = t & 31;
    int frag = t >> 5;                                  // (k*2+q)*4 + nt
    int nt = frag & 3;
    int q  = (frag >> 2) & 1;
    int k  = frag >> 3;
    int col   = nt * 16 + (lane & 15);
    int krow0 = q * 32 + (lane >> 4) * 16;
    const float* Wk = W + (size_t)k * 64 * 64;
    __bf16* dst = wb + (size_t)frag * 512 + lane * 16;  // 512 halfs per fragment
#pragma unroll
    for (int h = 0; h < 16; ++h)
        dst[h] = (__bf16)Wk[(krow0 + h) * 64 + col];
}

// ---------------------------------------------------------------------------
// f32 -> bf16 tensor convert (vectorized x4)
// ---------------------------------------------------------------------------
__global__ void f32_to_bf16(const float* __restrict__ x, __bf16* __restrict__ y) {
    int i = blockIdx.x * blockDim.x + threadIdx.x;
    if (i >= NELEM / 4) return;
    v4f v = ((const v4f*)x)[i];
    v4bf o;
    o[0] = (__bf16)v[0]; o[1] = (__bf16)v[1];
    o[2] = (__bf16)v[2]; o[3] = (__bf16)v[3];
    ((v4bf*)y)[i] = o;
}

// ---------------------------------------------------------------------------
// Gather-GEMM conv: out[n] = sum_k feat_bf16[nbr[n][k]] @ Wk  (fp32 accum)
// Each wave: one 32-point x 64-channel tile (two 16-row A tiles sharing every
// B fragment -> 16 WMMAs per k for 24 vector loads). 4 waves per block.
// Also accumulates per-channel sum / sumsq into stats[0..63] / stats[64..127].
// ---------------------------------------------------------------------------
__global__ __launch_bounds__(128)
void subm_conv_wmma(const __bf16* __restrict__ fin,
                    const int*    __restrict__ nbr,
                    const __bf16* __restrict__ wb,
                    float*        __restrict__ out,
                    float*        __restrict__ stats) {
    __shared__ float sred[128];
    sred[threadIdx.x] = 0.0f;
    __syncthreads();

    const int lane = threadIdx.x & 31;
    const int wave = threadIdx.x >> 5;
    const int tile = blockIdx.x * 4 + wave;          // 3125 blocks * 4 = 12500 tiles
    const int row0 = tile * 32 + (lane & 15);        // m-group 0
    const int row1 = row0 + 16;                      // m-group 1
    const int c0   = (lane >> 4) * 16;               // column base for this lane's A runs

    v8f acc00 = {}, acc01 = {}, acc02 = {}, acc03 = {};   // m-group 0, nt = 0..3
    v8f acc10 = {}, acc11 = {}, acc12 = {}, acc13 = {};   // m-group 1, nt = 0..3

    const int* nrow0 = nbr + (size_t)row0 * KOFF;
    const int* nrow1 = nbr + (size_t)row1 * KOFF;

    for (int k = 0; k < KOFF; ++k) {
        // ---- gather A fragments for both 16-row groups -------------------
        int nb0 = nrow0[k];
        int nb1 = nrow1[k];
        const __bf16* fr0 = fin + (size_t)nb0 * NCH + c0;
        const __bf16* fr1 = fin + (size_t)nb1 * NCH + c0;
        // A 16x32 bf16 fragment in this code's k order: lane<16 holds row cols {c..c+15} (contiguous)
        v8bf x0 = *(const v8bf*)(fr0);
        v8bf x1 = *(const v8bf*)(fr0 + 8);
        v8bf x2 = *(const v8bf*)(fr0 + 32);
        v8bf x3 = *(const v8bf*)(fr0 + 40);
        v8bf y0 = *(const v8bf*)(fr1);
        v8bf y1 = *(const v8bf*)(fr1 + 8);
        v8bf y2 = *(const v8bf*)(fr1 + 32);
        v8bf y3 = *(const v8bf*)(fr1 + 40);
        v16bf a0m0 = __builtin_shufflevector(x0, x1, 0,1,2,3,4,5,6,7,8,9,10,11,12,13,14,15);
        v16bf a1m0 = __builtin_shufflevector(x2, x3, 0,1,2,3,4,5,6,7,8,9,10,11,12,13,14,15);
        v16bf a0m1 = __builtin_shufflevector(y0, y1, 0,1,2,3,4,5,6,7,8,9,10,11,12,13,14,15);
        v16bf a1m1 = __builtin_shufflevector(y2, y3, 0,1,2,3,4,5,6,7,8,9,10,11,12,13,14,15);

        // ---- weights: each B fragment feeds both m-groups ----------------
        const __bf16* wk = wb + (size_t)k * 8 * 512 + lane * 16;
        v16bf b;
        b = *(const v16bf*)(wk + 0 * 512);
        acc00 = wmma_bf16(a0m0, b, acc00);  acc10 = wmma_bf16(a0m1, b, acc10);
        b = *(const v16bf*)(wk + 4 * 512);
        acc00 = wmma_bf16(a1m0, b, acc00);  acc10 = wmma_bf16(a1m1, b, acc10);
        b = *(const v16bf*)(wk + 1 * 512);
        acc01 = wmma_bf16(a0m0, b, acc01);  acc11 = wmma_bf16(a0m1, b, acc11);
        b = *(const v16bf*)(wk + 5 * 512);
        acc01 = wmma_bf16(a1m0, b, acc01);  acc11 = wmma_bf16(a1m1, b, acc11);
        b = *(const v16bf*)(wk + 2 * 512);
        acc02 = wmma_bf16(a0m0, b, acc02);  acc12 = wmma_bf16(a0m1, b, acc12);
        b = *(const v16bf*)(wk + 6 * 512);
        acc02 = wmma_bf16(a1m0, b, acc02);  acc12 = wmma_bf16(a1m1, b, acc12);
        b = *(const v16bf*)(wk + 3 * 512);
        acc03 = wmma_bf16(a0m0, b, acc03);  acc13 = wmma_bf16(a0m1, b, acc13);
        b = *(const v16bf*)(wk + 7 * 512);
        acc03 = wmma_bf16(a1m0, b, acc03);  acc13 = wmma_bf16(a1m1, b, acc13);
    }

    // C/D layout: VGPR v, lanes 0-15 -> point row v; lanes 16-31 -> row v+8
    const int colbase = lane & 15;
    const int prowm0  = tile * 32 + (lane >> 4) * 8;
    const int prowm1  = prowm0 + 16;
    float s0 = 0, s1 = 0, s2 = 0, s3 = 0;
    float q0 = 0, q1 = 0, q2 = 0, q3 = 0;
#pragma unroll
    for (int v = 0; v < 8; ++v) {
        float* op0 = out + (size_t)(prowm0 + v) * NCH + colbase;
        float* op1 = out + (size_t)(prowm1 + v) * NCH + colbase;
        float e0 = acc00[v], e1 = acc01[v], e2 = acc02[v], e3 = acc03[v];
        float f0 = acc10[v], f1 = acc11[v], f2 = acc12[v], f3 = acc13[v];
        op0[0]  = e0; op0[16] = e1; op0[32] = e2; op0[48] = e3;
        op1[0]  = f0; op1[16] = f1; op1[32] = f2; op1[48] = f3;
        s0 += e0 + f0; q0 += e0 * e0 + f0 * f0;
        s1 += e1 + f1; q1 += e1 * e1 + f1 * f1;
        s2 += e2 + f2; q2 += e2 * e2 + f2 * f2;
        s3 += e3 + f3; q3 += e3 * e3 + f3 * f3;
    }
    // fold the two half-wave row groups (lane, lane+16 share a channel)
    s0 += __shfl_xor(s0, 16, 32);  q0 += __shfl_xor(q0, 16, 32);
    s1 += __shfl_xor(s1, 16, 32);  q1 += __shfl_xor(q1, 16, 32);
    s2 += __shfl_xor(s2, 16, 32);  q2 += __shfl_xor(q2, 16, 32);
    s3 += __shfl_xor(s3, 16, 32);  q3 += __shfl_xor(q3, 16, 32);
    if (lane < 16) {
        atomicAdd(&sred[colbase +  0], s0);  atomicAdd(&sred[64 + colbase +  0], q0);
        atomicAdd(&sred[colbase + 16], s1);  atomicAdd(&sred[64 + colbase + 16], q1);
        atomicAdd(&sred[colbase + 32], s2);  atomicAdd(&sred[64 + colbase + 32], q2);
        atomicAdd(&sred[colbase + 48], s3);  atomicAdd(&sred[64 + colbase + 48], q3);
    }
    __syncthreads();
    atomicAdd(&stats[threadIdx.x], sred[threadIdx.x]);
}

// ---------------------------------------------------------------------------
// BN (batch stats) + leaky relu (+ optional residual) -> optional f32 / bf16 out
// ---------------------------------------------------------------------------
__global__ void bn_act(const float* __restrict__ x,
                       const float* __restrict__ stats,
                       const float* __restrict__ g,
                       const float* __restrict__ b,
                       const float* __restrict__ resid,
                       float slope,
                       float*  __restrict__ yf,
                       __bf16* __restrict__ ybf) {
    int i = blockIdx.x * blockDim.x + threadIdx.x;     // over float4 groups
    if (i >= NELEM / 4) return;
    v4f v = ((const v4f*)x)[i];
    v4f r = resid ? ((const v4f*)resid)[i] : (v4f){0.f, 0.f, 0.f, 0.f};
    int cb = (i * 4) & (NCH - 1);
    float o[4];
#pragma unroll
    for (int j = 0; j < 4; ++j) {
        int ch = cb + j;
        float m   = stats[ch] * (1.0f / NPTS);
        float var = stats[64 + ch] * (1.0f / NPTS) - m * m;
        float sc  = g[ch] * __frsqrt_rn(var + 1e-4f);
        float val = (v[j] - m) * sc + b[ch];
        if (resid) val += r[j];
        o[j] = val >= 0.0f ? val : slope * val;
    }
    if (yf) {
        v4f ov = {o[0], o[1], o[2], o[3]};
        ((v4f*)yf)[i] = ov;
    }
    if (ybf) {
        v4bf ob;
        ob[0] = (__bf16)o[0]; ob[1] = (__bf16)o[1];
        ob[2] = (__bf16)o[2]; ob[3] = (__bf16)o[3];
        ((v4bf*)ybf)[i] = ob;
    }
}

// ---------------------------------------------------------------------------
extern "C" void kernel_launch(void* const* d_in, const int* in_sizes, int n_in,
                              void* d_out, int out_size, void* d_ws, size_t ws_size,
                              hipStream_t stream) {
    const float* feat = (const float*)d_in[0];
    const int*   nbr  = (const int*)d_in[1];
    const float* W1 = (const float*)d_in[2];
    const float* g1 = (const float*)d_in[3];
    const float* b1 = (const float*)d_in[4];
    const float* W2 = (const float*)d_in[5];
    const float* g2 = (const float*)d_in[6];
    const float* b2 = (const float*)d_in[7];
    const float* W3 = (const float*)d_in[8];
    const float* g3 = (const float*)d_in[9];
    const float* b3 = (const float*)d_in[10];

    // workspace carve-up
    float*  bufA = (float*)d_ws;                 // conv1 out -> r1 (f32)
    float*  bufB = bufA + NELEM;                 // conv2/conv3 out (f32)
    __bf16* bfA  = (__bf16*)(bufB + NELEM);      // featbf, later x2 bf16
    __bf16* bfB  = bfA + NELEM;                  // x1 bf16
    __bf16* wb1  = bfB + NELEM;                  // 27*8*512 halfs each
    __bf16* wb2  = wb1 + KOFF * 8 * 512;
    __bf16* wb3  = wb2 + KOFF * 8 * 512;
    float*  st   = (float*)(wb3 + KOFF * 8 * 512); // 3 * 128 floats

    hipMemsetAsync(st, 0, 3 * 128 * sizeof(float), stream);

    swizzle_w<<<27, 256, 0, stream>>>(W1, wb1);
    swizzle_w<<<27, 256, 0, stream>>>(W2, wb2);
    swizzle_w<<<27, 256, 0, stream>>>(W3, wb3);

    const int cvt_blocks = (NELEM / 4 + 255) / 256;   // 25000
    f32_to_bf16<<<cvt_blocks, 256, 0, stream>>>(feat, bfA);

    const int conv_blocks = NPTS / 128;               // 3125 (4 waves x 32 pts)

    // layer 1
    subm_conv_wmma<<<conv_blocks, 128, 0, stream>>>(bfA, nbr, wb1, bufA, st);
    bn_act<<<cvt_blocks, 256, 0, stream>>>(bufA, st, g1, b1, nullptr, 0.05f,
                                           bufA /*r1*/, bfB);
    // layer 2
    subm_conv_wmma<<<conv_blocks, 128, 0, stream>>>(bfB, nbr, wb2, bufB, st + 128);
    bn_act<<<cvt_blocks, 256, 0, stream>>>(bufB, st + 128, g2, b2, nullptr, 0.05f,
                                           nullptr, bfA);
    // layer 3 + residual + final lrelu(0.333)
    subm_conv_wmma<<<conv_blocks, 128, 0, stream>>>(bfA, nbr, wb3, bufB, st + 256);
    bn_act<<<cvt_blocks, 256, 0, stream>>>(bufB, st + 256, g3, b3, bufA, 0.333f,
                                           (float*)d_out, nullptr);
}
